// DecoderLayer_80504866997021
// MI455X (gfx1250) — compile-verified, matrix operand order checked by reading, not run on hardware
//
#include <hip/hip_runtime.h>
#include <hip/hip_bf16.h>

// ---------------------------------------------------------------------------
// BEVFormer-style decoder layer for MI455X (gfx1250, wave32, WMMA).
// All GEMMs run through v_wmma_f32_16x16x32_f16 (f16 inputs, f32 accum).
// Softmax row reductions use DPP16 butterflies (VALU) instead of LDS permutes.
// ---------------------------------------------------------------------------

typedef _Float16 v16h __attribute__((ext_vector_type(16)));
typedef _Float16 v8h  __attribute__((ext_vector_type(8)));
typedef _Float16 v4h  __attribute__((ext_vector_type(4)));
typedef float    v8f  __attribute__((ext_vector_type(8)));

#define D_MODEL 256
#define N_HEADS 8
#define N_POINTS 8
#define DIM_FF 1024
#define BQ 4
#define LQ 2048
#define BEV 128
#define DH 32
#define TOK (BQ*LQ)          // 8192 query tokens
#define VTOK (BQ*BEV*BEV)    // 65536 value tokens

__device__ __forceinline__ v8f vzero8() {
  v8f z;
#pragma unroll
  for (int i = 0; i < 8; ++i) z[i] = 0.0f;
  return z;
}

__device__ __forceinline__ v8f wmma32(v16h a, v16h b, v8f c) {
  // D = A(16x32 f16) * B(32x16 f16) + C(16x16 f32)
  return __builtin_amdgcn_wmma_f32_16x16x32_f16(
      /*neg_a=*/false, a, /*neg_b=*/false, b,
      /*c_mod=*/(short)0, c, /*reuse_a=*/false, /*reuse_b=*/false);
}

// --- DPP16 half-wave butterflies (no LDS traffic, co-issue with VALU) ------
// Stages: quad_perm(1,0,3,2)=xor1, quad_perm(2,3,0,1)=xor2,
// row_half_mirror(0x141)~xor4, row_mirror(0x140)~xor8 (valid because earlier
// stages make quads/octets uniform).  Reduction stays within each 16-lane half.
template <int CTRL>
__device__ __forceinline__ float dpp_movf(float x) {
  return __int_as_float(__builtin_amdgcn_update_dpp(
      0, __float_as_int(x), CTRL, 0xF, 0xF, true));
}
__device__ __forceinline__ float red_max16(float x) {
  x = fmaxf(x, dpp_movf<0xB1>(x));
  x = fmaxf(x, dpp_movf<0x4E>(x));
  x = fmaxf(x, dpp_movf<0x141>(x));
  x = fmaxf(x, dpp_movf<0x140>(x));
  return x;
}
__device__ __forceinline__ float red_sum16(float x) {
  x += dpp_movf<0xB1>(x);
  x += dpp_movf<0x4E>(x);
  x += dpp_movf<0x141>(x);
  x += dpp_movf<0x140>(x);
  return x;
}

// Load a 16x32 f16 fragment (A-layout == B-layout for NT-stored operands):
// lane L reads row (L&15), 8 halves at col k0 and 8 halves at col k0+16,
// with k0 = (L<16) ? 0 : 8.  `ld` = row stride in halves.
__device__ __forceinline__ v16h load_frag(const _Float16* base, int ld) {
  const int lane = threadIdx.x & 31;
  const _Float16* p = base + (size_t)(lane & 15) * ld + ((lane & 16) ? 8 : 0);
  v8h lo = *(const v8h*)(p);
  v8h hi = *(const v8h*)(p + 16);
  return __builtin_shufflevector(lo, hi, 0,1,2,3,4,5,6,7,8,9,10,11,12,13,14,15);
}

// ---------------------------------------------------------------------------
// Generic WMMA GEMM:  C[M,N] = A[M,K] * W[N,K]^T + bias  (+relu) (+resid)
// Block tile 128(M) x 64(N) x 32(K); 8 waves, each wave a 32x32 sub-tile.
// A and W are f32 in HBM, converted to f16 while staging into LDS.
// RELU / RESID are compile-time so the epilogue is branch-free.
// ---------------------------------------------------------------------------
template <bool RELU, bool RESID>
__global__ void __launch_bounds__(256)
gemm_wmma(const float* __restrict__ A, const float* __restrict__ W,
          const float* __restrict__ bias, const float* __restrict__ resid,
          float* __restrict__ C, int M, int N, int K) {
  __shared__ __align__(16) _Float16 sA[128 * 32];
  __shared__ __align__(16) _Float16 sB[64 * 32];

  const int tid  = threadIdx.x;
  const int lane = tid & 31;
  const int wave = tid >> 5;
  const int wm   = wave >> 1;   // 0..3 -> M sub-tile
  const int wn   = wave & 1;    // 0..1 -> N sub-tile
  const int m0   = blockIdx.y * 128;
  const int n0   = blockIdx.x * 64;

  v8f acc00 = vzero8(), acc01 = vzero8(), acc10 = vzero8(), acc11 = vzero8();

  const int ar = tid >> 1;             // 0..127
  const int ac = (tid & 1) * 16;       // 0 or 16
  const int br = tid >> 2;             // 0..63
  const int bc = (tid & 3) * 8;        // 0,8,16,24

  const float* aRow = A + (size_t)(m0 + ar) * K + ac;
  const float* bRow = W + (size_t)(n0 + br) * K + bc;

  for (int kb = 0; kb < K; kb += 32) {
    __syncthreads();  // previous iteration's fragment reads done
    {
      const float4* ap = (const float4*)(aRow + kb);
#pragma unroll
      for (int j = 0; j < 4; ++j) {
        float4 f = ap[j];
        v4h h = {(_Float16)f.x, (_Float16)f.y, (_Float16)f.z, (_Float16)f.w};
        *(v4h*)(&sA[ar * 32 + ac + j * 4]) = h;
      }
      const float4* bp = (const float4*)(bRow + kb);
#pragma unroll
      for (int j = 0; j < 2; ++j) {
        float4 f = bp[j];
        v4h h = {(_Float16)f.x, (_Float16)f.y, (_Float16)f.z, (_Float16)f.w};
        *(v4h*)(&sB[br * 32 + bc + j * 4]) = h;
      }
      // Warm L2/WGP$ one k-tile ahead (global_prefetch_b8 on gfx1250).
      if (kb + 32 < K) {
        __builtin_prefetch(aRow + kb + 32, 0, 3);
        __builtin_prefetch(bRow + kb + 32, 0, 3);
      }
    }
    __syncthreads();

    v16h a0 = load_frag(sA + (wm * 32) * 32, 32);
    v16h a1 = load_frag(sA + (wm * 32 + 16) * 32, 32);
    v16h b0 = load_frag(sB + (wn * 32) * 32, 32);
    v16h b1 = load_frag(sB + (wn * 32 + 16) * 32, 32);

    acc00 = wmma32(a0, b0, acc00);
    acc01 = wmma32(a0, b1, acc01);
    acc10 = wmma32(a1, b0, acc10);
    acc11 = wmma32(a1, b1, acc11);
  }

  // Epilogue. C layout: VGPR r -> row r (lanes 0-15) / row r+8 (lanes 16-31),
  // col = lane&15 within the 16-wide N tile.  Pointer-stepped stores.
  const int hi = (lane & 16) ? 8 : 0;
#pragma unroll
  for (int im = 0; im < 2; ++im) {
#pragma unroll
    for (int jn = 0; jn < 2; ++jn) {
      v8f a = im == 0 ? (jn == 0 ? acc00 : acc01) : (jn == 0 ? acc10 : acc11);
      const int n = n0 + wn * 32 + jn * 16 + (lane & 15);
      const float bv = bias[n];
      const size_t rbase = (size_t)(m0 + wm * 32 + im * 16 + hi);
      float* cp = C + rbase * N + n;
      const float* rp = RESID ? (resid + rbase * N + n) : nullptr;
#pragma unroll
      for (int r = 0; r < 8; ++r) {
        float v = a[r] + bv;
        if (RELU) v = fmaxf(v, 0.0f);
        if (RESID) v += rp[(size_t)r * N];
        cp[(size_t)r * N] = v;
      }
    }
  }
}

// ---------------------------------------------------------------------------
// Flash attention: one wave per (b, h, 16-query tile). Q pre-scaled by 1/sqrt(DH).
// qh,kh: f16 [B,H,LQ,DH]; vt: f16 [B,H,DH,LQ]; o: f32 [B,LQ,D_MODEL].
// ---------------------------------------------------------------------------
__global__ void __launch_bounds__(256)
attn_flash(const _Float16* __restrict__ qh, const _Float16* __restrict__ kh,
           const _Float16* __restrict__ vt, float* __restrict__ o) {
  __shared__ __align__(16) _Float16 sP[8 * 16 * 32];  // per-wave 16x32 P tile

  const int wave = threadIdx.x >> 5;
  const int lane = threadIdx.x & 31;
  const int gw = blockIdx.x * 8 + wave;   // 0..4095
  const int qt = gw & 127;                // LQ/16 tiles
  const int bh = gw >> 7;                 // b*8 + h
  const int b = bh >> 3, h = bh & 7;

  const _Float16* Q  = qh + ((size_t)bh * LQ + qt * 16) * DH;
  const _Float16* Kb = kh + (size_t)bh * LQ * DH;
  const _Float16* Vt = vt + (size_t)bh * DH * LQ;
  _Float16* P = &sP[wave * 16 * 32];

  const v16h qf = load_frag(Q, DH);
  v8f o0 = vzero8(), o1 = vzero8();
  float m[8], l[8];
#pragma unroll
  for (int r = 0; r < 8; ++r) { m[r] = -1e30f; l[r] = 0.0f; }

  const int hi = (lane & 16) ? 8 : 0;

  for (int kt = 0; kt < LQ; kt += 32) {
    v16h k0f = load_frag(Kb + (size_t)kt * DH, DH);
    v16h k1f = load_frag(Kb + (size_t)(kt + 16) * DH, DH);
    v8f s0 = wmma32(qf, k0f, vzero8());
    v8f s1 = wmma32(qf, k1f, vzero8());

#pragma unroll
    for (int r = 0; r < 8; ++r) {
      float a0 = s0[r], a1 = s1[r];
      const float mt = red_max16(fmaxf(a0, a1));   // DPP16 butterfly, no LDS
      const float mn = fmaxf(m[r], mt);
      const float sc = __expf(m[r] - mn);
      const float p0 = __expf(a0 - mn);
      const float p1 = __expf(a1 - mn);
      const float ps = red_sum16(p0 + p1);         // DPP16 butterfly, no LDS
      l[r] = l[r] * sc + ps;
      m[r] = mn;
      o0[r] *= sc;
      o1[r] *= sc;
      const int row = r + hi;
      P[row * 32 + (lane & 15)]      = (_Float16)p0;
      P[row * 32 + 16 + (lane & 15)] = (_Float16)p1;
    }
    asm volatile("s_wait_dscnt 0" ::: "memory");  // P writes visible to this wave

    v16h pf  = load_frag(P, 32);
    v16h v0f = load_frag(Vt + kt, LQ);
    v16h v1f = load_frag(Vt + (size_t)16 * LQ + kt, LQ);
    o0 = wmma32(pf, v0f, o0);
    o1 = wmma32(pf, v1f, o1);
  }

#pragma unroll
  for (int r = 0; r < 8; ++r) {
    const float inv = 1.0f / l[r];
    const size_t tok = (size_t)b * LQ + qt * 16 + r + hi;
    o[tok * D_MODEL + h * DH + (lane & 15)]      = o0[r] * inv;
    o[tok * D_MODEL + h * DH + 16 + (lane & 15)] = o1[r] * inv;
  }
}

// ---------------------------------------------------------------------------
// Elementwise / reshape kernels
// ---------------------------------------------------------------------------
__global__ void add_kernel(const float* a, const float* b, float* c, int n) {
  int i = blockIdx.x * 256 + threadIdx.x;
  if (i < n) c[i] = a[i] + b[i];
}

// [TOK, 256] f32 -> f16 [B,H,LQ,DH] (optionally pre-scaled, for Q)
__global__ void split_heads(const float* __restrict__ in, _Float16* __restrict__ out,
                            float scale) {
  const size_t i = (size_t)blockIdx.x * 256 + threadIdx.x;  // TOK*256
  const int c = (int)(i & 255);
  const size_t t = i >> 8;
  const int h = c >> 5, d = c & 31;
  const size_t b = t >> 11, lq = t & 2047;
  out[(((b * N_HEADS + h) * LQ) + lq) * DH + d] = (_Float16)(in[i] * scale);
}

// [TOK, 256] f32 -> f16 [B,H,DH,LQ] (transposed V for the P*V WMMA)
__global__ void split_vT(const float* __restrict__ in, _Float16* __restrict__ out) {
  const size_t i = (size_t)blockIdx.x * 256 + threadIdx.x;
  const int c = (int)(i & 255);
  const size_t t = i >> 8;
  const int h = c >> 5, d = c & 31;
  const size_t b = t >> 11, lq = t & 2047;
  out[(((b * N_HEADS + h) * DH) + d) * LQ + lq] = (_Float16)in[i];
}

// Wave-per-row layernorm over 256 columns.
__global__ void __launch_bounds__(256)
layernorm_rows(const float* __restrict__ x, const float* __restrict__ g,
               const float* __restrict__ bta, float* __restrict__ y) {
  const int wave = threadIdx.x >> 5, lane = threadIdx.x & 31;
  const size_t row = (size_t)blockIdx.x * 8 + wave;
  const float* xr = x + row * D_MODEL;
  float v[8];
  float s = 0.0f, s2 = 0.0f;
#pragma unroll
  for (int i = 0; i < 8; ++i) {
    v[i] = xr[lane * 8 + i];
    s += v[i];
    s2 += v[i] * v[i];
  }
  // 4 DPP16 stages + one cross-half exchange.
  s = red_sum16(s);
  s2 = red_sum16(s2);
  s += __shfl_xor(s, 16);
  s2 += __shfl_xor(s2, 16);
  const float mu = s * (1.0f / D_MODEL);
  const float var = s2 * (1.0f / D_MODEL) - mu * mu;
  const float r = rsqrtf(var + 1e-5f);
#pragma unroll
  for (int i = 0; i < 8; ++i) {
    const int c = lane * 8 + i;
    y[row * D_MODEL + c] = (v[i] - mu) * r * g[c] + bta[c];
  }
}

// ---------------------------------------------------------------------------
// Deformable-attention bilinear sampling.
// One thread per (b, lq, h, d).  vproj: f32 [B, BEV*BEV, 256].
// off: [TOK,128] (h*16 + p*2 + c), aw: [TOK,64] (h*8 + p, pre-softmax).
// ---------------------------------------------------------------------------
__global__ void deform_sample(const float* __restrict__ vproj,
                              const float* __restrict__ ref2d,
                              const float* __restrict__ off,
                              const float* __restrict__ aw,
                              float* __restrict__ samp) {
  const size_t i = (size_t)blockIdx.x * 256 + threadIdx.x;  // TOK*256
  const int c = (int)(i & 255);
  const size_t t = i >> 8;
  const int h = c >> 5, d = c & 31;
  const int b = (int)(t >> 11);

  const float rx = ref2d[t * 2 + 0];
  const float ry = ref2d[t * 2 + 1];
  const float* ap = aw + t * 64 + h * 8;
  const float* op = off + t * 128 + h * 16;

  // softmax over the 8 points
  float mx = -1e30f;
#pragma unroll
  for (int p = 0; p < N_POINTS; ++p) mx = fmaxf(mx, ap[p]);
  float e[8], s = 0.0f;
#pragma unroll
  for (int p = 0; p < N_POINTS; ++p) { e[p] = __expf(ap[p] - mx); s += e[p]; }
  const float inv = 1.0f / s;

  const float* vb = vproj + (size_t)b * (BEV * BEV) * D_MODEL + h * DH + d;
  float acc = 0.0f;
#pragma unroll
  for (int p = 0; p < N_POINTS; ++p) {
    const float lx = rx + op[2 * p]     * (1.0f / BEV);
    const float ly = ry + op[2 * p + 1] * (1.0f / BEV);
    const float x = lx * (float)BEV - 0.5f;
    const float y = ly * (float)BEV - 0.5f;
    const float x0 = floorf(x), y0 = floorf(y);
    const float fx = x - x0, fy = y - y0;
    const int xi = (int)x0, yi = (int)y0;

    float v4[4];
#pragma unroll
    for (int k = 0; k < 4; ++k) {
      const int yy = yi + (k >> 1);
      const int xx = xi + (k & 1);
      const bool valid = (yy >= 0) && (yy < BEV) && (xx >= 0) && (xx < BEV);
      const int yc = min(max(yy, 0), BEV - 1);
      const int xc = min(max(xx, 0), BEV - 1);
      const float g = vb[(size_t)(yc * BEV + xc) * D_MODEL];
      v4[k] = valid ? g : 0.0f;
    }
    const float bil = v4[0] * (1.0f - fx) * (1.0f - fy) + v4[1] * fx * (1.0f - fy) +
                      v4[2] * (1.0f - fx) * fy + v4[3] * fx * fy;
    acc += bil * e[p] * inv;
  }
  samp[i] = acc;
}

// ---------------------------------------------------------------------------
// Host orchestration
// ---------------------------------------------------------------------------
extern "C" void kernel_launch(void* const* d_in, const int* in_sizes, int n_in,
                              void* d_out, int out_size, void* d_ws, size_t ws_size,
                              hipStream_t stream) {
  const float* query      = (const float*)d_in[0];
  const float* value      = (const float*)d_in[1];
  const float* query_pos  = (const float*)d_in[2];
  const float* ref_2d     = (const float*)d_in[3];
  const float* in_proj_w  = (const float*)d_in[4];
  const float* in_proj_b  = (const float*)d_in[5];
  const float* out_proj_w = (const float*)d_in[6];
  const float* out_proj_b = (const float*)d_in[7];
  const float* norm1_g    = (const float*)d_in[8];
  const float* norm1_b    = (const float*)d_in[9];
  const float* vproj_w    = (const float*)d_in[10];
  const float* vproj_b    = (const float*)d_in[11];
  const float* off_w      = (const float*)d_in[12];
  const float* off_b      = (const float*)d_in[13];
  const float* aw_w       = (const float*)d_in[14];
  const float* aw_b       = (const float*)d_in[15];
  const float* outp_w     = (const float*)d_in[16];
  const float* outp_b     = (const float*)d_in[17];
  const float* norm2_g    = (const float*)d_in[18];
  const float* norm2_b    = (const float*)d_in[19];
  const float* lin1_w     = (const float*)d_in[20];
  const float* lin1_b     = (const float*)d_in[21];
  const float* lin2_w     = (const float*)d_in[22];
  const float* lin2_b     = (const float*)d_in[23];
  const float* norm3_g    = (const float*)d_in[24];
  const float* norm3_b    = (const float*)d_in[25];

  const size_t MB = 1ull << 20;
  char* ws = (char*)d_ws;
  // Workspace layout (with deliberate lifetime-based reuse; peak 114 MB):
  float*    x1   = (float*)(ws + 0);        // 8MB  x1 -> o_heads -> samp
  float*    bufA = (float*)(ws + 8 * MB);   // 8MB  q_lin -> q2n -> q4
  float*    bufB = (float*)(ws + 16 * MB);  // 8MB  k_lin -> q_res -> q5
  float*    bufC = (float*)(ws + 24 * MB);  // 8MB  v_lin -> q3
  _Float16* qh   = (_Float16*)(ws + 32 * MB);  // 4MB
  _Float16* kh   = (_Float16*)(ws + 36 * MB);  // 4MB
  _Float16* vt   = (_Float16*)(ws + 40 * MB);  // 4MB
  float*    offb = (float*)(ws + 44 * MB);  // 4MB  [TOK,128]
  float*    awb  = (float*)(ws + 48 * MB);  // 2MB  [TOK,64]
  float*    big  = (float*)(ws + 50 * MB);  // 64MB vproj -> h1 (32MB)

  const dim3 blk(256);
  const dim3 g_tok(TOK * D_MODEL / 256);        // 8192 blocks, element-wise over tokens
  const float qscale = 0.17677669529663687f;    // 1/sqrt(DH)

  // 1) x1 = query + query_pos
  add_kernel<<<g_tok, blk, 0, stream>>>(query, query_pos, x1, TOK * D_MODEL);

  // 2-4) QKV projections (q from x1, k/v from query)
  gemm_wmma<false, false><<<dim3(256 / 64, TOK / 128), blk, 0, stream>>>(
      x1, in_proj_w, in_proj_b, nullptr, bufA, TOK, 256, 256);
  gemm_wmma<false, false><<<dim3(256 / 64, TOK / 128), blk, 0, stream>>>(
      query, in_proj_w + 256 * 256, in_proj_b + 256, nullptr, bufB, TOK, 256, 256);
  gemm_wmma<false, false><<<dim3(256 / 64, TOK / 128), blk, 0, stream>>>(
      query, in_proj_w + 512 * 256, in_proj_b + 512, nullptr, bufC, TOK, 256, 256);

  // 5-7) head split (+scale into Q) and V transpose, f32 -> f16
  split_heads<<<g_tok, blk, 0, stream>>>(bufA, qh, qscale);
  split_heads<<<g_tok, blk, 0, stream>>>(bufB, kh, 1.0f);
  split_vT<<<g_tok, blk, 0, stream>>>(bufC, vt);

  // 8) flash attention -> o_heads (reuse x1)
  attn_flash<<<dim3(4096 / 8), blk, 0, stream>>>(qh, kh, vt, x1);

  // 9) out-proj + residual(query) -> q_res (bufB)
  gemm_wmma<false, true><<<dim3(256 / 64, TOK / 128), blk, 0, stream>>>(
      x1, out_proj_w, out_proj_b, query, bufB, TOK, 256, 256);

  // 10) LN1 -> q2n (bufA)
  layernorm_rows<<<dim3(TOK / 8), blk, 0, stream>>>(bufB, norm1_g, norm1_b, bufA);

  // 11) value projection -> big
  gemm_wmma<false, false><<<dim3(256 / 64, VTOK / 128), blk, 0, stream>>>(
      value, vproj_w, vproj_b, nullptr, big, VTOK, 256, 256);

  // 12-13) sampling offsets / attention weights from q2n
  gemm_wmma<false, false><<<dim3(128 / 64, TOK / 128), blk, 0, stream>>>(
      bufA, off_w, off_b, nullptr, offb, TOK, 128, 256);
  gemm_wmma<false, false><<<dim3(64 / 64, TOK / 128), blk, 0, stream>>>(
      bufA, aw_w, aw_b, nullptr, awb, TOK, 64, 256);

  // 14) bilinear sampling -> samp (reuse x1)
  deform_sample<<<g_tok, blk, 0, stream>>>(big, ref_2d, offb, awb, x1);

  // 15) output projection + residual(q_res) -> q3 (bufC)
  gemm_wmma<false, true><<<dim3(256 / 64, TOK / 128), blk, 0, stream>>>(
      x1, outp_w, outp_b, bufB, bufC, TOK, 256, 256);

  // 16) LN2 -> q4 (bufA)
  layernorm_rows<<<dim3(TOK / 8), blk, 0, stream>>>(bufC, norm2_g, norm2_b, bufA);

  // 17) FFN lin1 (+relu) -> h1 (big)
  gemm_wmma<true, false><<<dim3(DIM_FF / 64, TOK / 128), blk, 0, stream>>>(
      bufA, lin1_w, lin1_b, nullptr, big, TOK, DIM_FF, 256);

  // 18) FFN lin2 + residual(q4) -> q5 (bufB)
  gemm_wmma<false, true><<<dim3(256 / 64, TOK / 128), blk, 0, stream>>>(
      big, lin2_w, lin2_b, bufA, bufB, TOK, 256, DIM_FF);

  // 19) LN3 -> d_out
  layernorm_rows<<<dim3(TOK / 8), blk, 0, stream>>>(bufB, norm3_g, norm3_b, (float*)d_out);
}
